// ChimeraMambaKANBlock_89756226551981
// MI455X (gfx1250) — matrix operand order repaired and checked by reading, not compiled, not run on hardware
//
#include <hip/hip_runtime.h>
#include <hip/hip_bf16.h>
#include <stdint.h>

// ---------------- problem constants (fixed by reference) ----------------
#define DIMX     512
#define D_STATE  16
#define D_CONVX  4
#define D_INNER  1024
#define DT_RANK  32
#define NGRIDS   8
#define SEQ_B    2
#define SEQ_L    2048
#define NROWS    (SEQ_B * SEQ_L)          // 4096 tokens
#define INV_DEN  (1.0f / 0.33f)

typedef __attribute__((ext_vector_type(16))) __bf16        bf16x16;
typedef __attribute__((ext_vector_type(8)))  float         f32x8;
typedef __attribute__((ext_vector_type(16))) unsigned short u16x16;

__device__ __forceinline__ unsigned short f2bf(float f) {
  unsigned int u = __builtin_bit_cast(unsigned int, f);
  u += 0x7fffu + ((u >> 16) & 1u);            // round-to-nearest-even
  return (unsigned short)(u >> 16);
}

// ======================================================================
// bf16 WMMA GEMM:  out[M x N] = act(A[M x K] * W[N x K]^T [+ bias]) [+ res]
// Compile-time flags -> branch-free, guard-free epilogue. All launches use
// exact tiling: M % 128 == 0, N % 64 == 0 (N=64 handled by one block col).
// Block = 128 threads = 4 wave32 waves (2x2); wave tile 64M x 32N.
// Weights are L2-resident on MI455X (7.5MB bf16 vs 192MB L2), so per-wave
// fragment loads straight from global are L2 hits; contiguous k = 16*lh + j operands.
// ======================================================================
template <int ACT, bool HAS_BIAS, bool HAS_RES, bool HAS_BF>
__global__ __launch_bounds__(128) void gemm_bf16_wmma(
    const unsigned short* __restrict__ A, int lda,
    const unsigned short* __restrict__ W, int ldw,
    const float* __restrict__ bias,
    const float* __restrict__ residual,     // ld = N
    float* __restrict__ out,                // ld = N
    unsigned short* __restrict__ out_bf,    // ld = N
    int N, int K)
{
  const int wave = threadIdx.x >> 5;
  const int lane = threadIdx.x & 31;
  const int lm   = lane & 15;
  const int lh   = lane >> 4;
  const int m0   = blockIdx.y * 128 + (wave >> 1) * 64;
  const int n0   = blockIdx.x * 64  + (wave & 1)  * 32;

  f32x8 acc[4][2];
  #pragma unroll
  for (int i = 0; i < 4; ++i) { acc[i][0] = {}; acc[i][1] = {}; }

  for (int k0 = 0; k0 < K; k0 += 32) {
    bf16x16 af[4], wf[2];
    // A fragment: 16x32 bf16; lane row m=lm, K = 16*lh + j
    #pragma unroll
    for (int i = 0; i < 4; ++i) {
      const unsigned short* ap = A + (size_t)(m0 + 16 * i + lm) * lda + k0;
      __builtin_prefetch((const void*)(ap + 32), 0, 1);   // global_prefetch next K block
      u16x16 t;
      #pragma unroll
      for (int j = 0; j < 8; ++j) {
        t[j]     = ap[16 * lh + j];
        t[8 + j] = ap[16 * lh + 8 + j];
      }
      af[i] = __builtin_bit_cast(bf16x16, t);
    }
    // B fragment: 32x16 bf16; lane col n=lm, K = 16*lh + j
    #pragma unroll
    for (int jn = 0; jn < 2; ++jn) {
      const unsigned short* wp = W + (size_t)(n0 + 16 * jn + lm) * ldw + k0 + 16 * lh;
      u16x16 t;
      #pragma unroll
      for (int j = 0; j < 16; ++j) t[j] = wp[j];
      wf[jn] = __builtin_bit_cast(bf16x16, t);
    }
    #pragma unroll
    for (int i = 0; i < 4; ++i)
      #pragma unroll
      for (int jn = 0; jn < 2; ++jn)
        acc[i][jn] = __builtin_amdgcn_wmma_f32_16x16x32_bf16(
            false, af[i], false, wf[jn], (short)0, acc[i][jn], false, false);
  }

  // Epilogue. C/D layout: VGPR r holds row m = r + 8*lh, col n = lm.
  #pragma unroll
  for (int jn = 0; jn < 2; ++jn) {
    const int col = n0 + 16 * jn + lm;
    const float bv = HAS_BIAS ? bias[col] : 0.f;
    #pragma unroll
    for (int i = 0; i < 4; ++i) {
      #pragma unroll
      for (int r = 0; r < 8; ++r) {
        const size_t idx = (size_t)(m0 + 16 * i + 8 * lh + r) * N + col;
        float v = acc[i][jn][r];
        if (HAS_BIAS) v += bv;
        if (ACT == 1) v = (v > 20.f) ? v : __logf(1.f + __expf(v));   // softplus
        if (HAS_RES)  v += residual[idx];
        out[idx] = v;
        if (HAS_BF) out_bf[idx] = f2bf(v);
      }
    }
  }
}

// ---------------- block-wide sum over 128 threads ----------------
__device__ __forceinline__ float block_sum128(float v, float* sm) {
  const int t = threadIdx.x;
  sm[t] = v;
  __syncthreads();
  #pragma unroll
  for (int s = 64; s > 0; s >>= 1) {
    if (t < s) sm[t] += sm[t + s];
    __syncthreads();
  }
  float r = sm[0];
  __syncthreads();
  return r;
}

// u = LN(LN(x, w1,b1), w2,b2)  -> bf16 ; one block (128 thr) per token row
__global__ __launch_bounds__(128) void double_ln_bf16_kernel(
    const float* __restrict__ x,
    const float* __restrict__ w1, const float* __restrict__ b1,
    const float* __restrict__ w2, const float* __restrict__ b2,
    unsigned short* __restrict__ out)
{
  __shared__ float sm[128];
  const int row = blockIdx.x;
  const int t   = threadIdx.x;
  const float* xr = x + (size_t)row * DIMX;
  float v[4], s = 0.f;
  #pragma unroll
  for (int i = 0; i < 4; ++i) { v[i] = xr[t + 128 * i]; s += v[i]; }
  s = block_sum128(s, sm);
  const float mu = s * (1.f / DIMX);
  float sq = 0.f;
  #pragma unroll
  for (int i = 0; i < 4; ++i) { float d = v[i] - mu; sq += d * d; }
  sq = block_sum128(sq, sm);
  const float rs = rsqrtf(sq * (1.f / DIMX) + 1e-5f);
  float y[4], s2 = 0.f;
  #pragma unroll
  for (int i = 0; i < 4; ++i) {
    int c = t + 128 * i;
    y[i] = (v[i] - mu) * rs * w1[c] + b1[c];
    s2 += y[i];
  }
  s2 = block_sum128(s2, sm);
  const float mu2 = s2 * (1.f / DIMX);
  float sq2 = 0.f;
  #pragma unroll
  for (int i = 0; i < 4; ++i) { float d = y[i] - mu2; sq2 += d * d; }
  sq2 = block_sum128(sq2, sm);
  const float rs2 = rsqrtf(sq2 * (1.f / DIMX) + 1e-5f);
  #pragma unroll
  for (int i = 0; i < 4; ++i) {
    int c = t + 128 * i;
    out[(size_t)row * DIMX + c] = f2bf((y[i] - mu2) * rs2 * w2[c] + b2[c]);
  }
}

// Double LN + FasterKAN basis: basis[row, c*8+g] = 1 - tanh^2((k[c]-grid[g])/0.33)
// 8 bf16 per channel packed into one 16-byte store (b128).
__global__ __launch_bounds__(128) void kan_basis_kernel(
    const float* __restrict__ x,
    const float* __restrict__ w1, const float* __restrict__ b1,
    const float* __restrict__ w2, const float* __restrict__ b2,
    const float* __restrict__ gridv,
    unsigned short* __restrict__ basis)
{
  __shared__ float sm[128];
  const int row = blockIdx.x;
  const int t   = threadIdx.x;
  const float* xr = x + (size_t)row * DIMX;
  float v[4], s = 0.f;
  #pragma unroll
  for (int i = 0; i < 4; ++i) { v[i] = xr[t + 128 * i]; s += v[i]; }
  s = block_sum128(s, sm);
  const float mu = s * (1.f / DIMX);
  float sq = 0.f;
  #pragma unroll
  for (int i = 0; i < 4; ++i) { float d = v[i] - mu; sq += d * d; }
  sq = block_sum128(sq, sm);
  const float rs = rsqrtf(sq * (1.f / DIMX) + 1e-5f);
  float y[4], s2 = 0.f;
  #pragma unroll
  for (int i = 0; i < 4; ++i) {
    int c = t + 128 * i;
    y[i] = (v[i] - mu) * rs * w1[c] + b1[c];
    s2 += y[i];
  }
  s2 = block_sum128(s2, sm);
  const float mu2 = s2 * (1.f / DIMX);
  float sq2 = 0.f;
  #pragma unroll
  for (int i = 0; i < 4; ++i) { float d = y[i] - mu2; sq2 += d * d; }
  sq2 = block_sum128(sq2, sm);
  const float rs2 = rsqrtf(sq2 * (1.f / DIMX) + 1e-5f);
  float gr[NGRIDS];
  #pragma unroll
  for (int g = 0; g < NGRIDS; ++g) gr[g] = gridv[g];
  #pragma unroll
  for (int i = 0; i < 4; ++i) {
    int c = t + 128 * i;
    float kv = (y[i] - mu2) * rs2 * w2[c] + b2[c];
    union { unsigned short us[8]; uint4 v4; } pk;
    #pragma unroll
    for (int g = 0; g < NGRIDS; ++g) {
      float tt = tanhf((kv - gr[g]) * INV_DEN);
      pk.us[g] = f2bf(1.f - tt * tt);
    }
    *(uint4*)(basis + (size_t)row * (DIMX * NGRIDS) + (size_t)c * NGRIDS) = pk.v4;
  }
}

// causal depthwise conv (k=4) + bias + SiLU; reads xm = xz[:, :D_INNER]
__global__ __launch_bounds__(256) void conv_silu_kernel(
    const float* __restrict__ xz, const float* __restrict__ cw,
    const float* __restrict__ cb, float* __restrict__ xc,
    unsigned short* __restrict__ xcbf)
{
  int i = blockIdx.x * 256 + threadIdx.x;
  if (i >= NROWS * D_INNER) return;
  int c   = i & (D_INNER - 1);
  int row = i >> 10;
  int l   = row & (SEQ_L - 1);
  int b   = row >> 11;
  float s = cb[c];
  #pragma unroll
  for (int t = 0; t < D_CONVX; ++t) {
    int ls = l - (D_CONVX - 1) + t;
    if (ls >= 0)
      s += cw[c * D_CONVX + t] * xz[(size_t)(b * SEQ_L + ls) * (2 * D_INNER) + c];
  }
  float sig = 1.f / (1.f + __expf(-s));
  float vv  = s * sig;
  xc[i]   = vv;
  xcbf[i] = f2bf(vv);
}

// Selective scan: one lane per (b, d, n); 16-lane shuffle reduction per step.
// y[b,l,d] = sum_n h[n]*C[n] + D[d]*xc  with  h = exp(delta*A)*h + delta*B*xc
__global__ __launch_bounds__(256) void ssm_scan_kernel(
    const float* __restrict__ delta, const float* __restrict__ dbc,
    const float* __restrict__ xc, const float* __restrict__ A_log,
    const float* __restrict__ D_param, float* __restrict__ y)
{
  const int gid = blockIdx.x * 256 + threadIdx.x;     // 0 .. 32767
  const int n = gid & (D_STATE - 1);
  const int d = (gid >> 4) & (D_INNER - 1);
  const int b = gid >> 14;
  const float a  = -__expf(A_log[d * D_STATE + n]);
  const float Dp = D_param[d];
  float h = 0.f;
  const size_t base = (size_t)b * SEQ_L;
  for (int l = 0; l < SEQ_L; ++l) {
    const size_t r = base + l;
    float dlt = delta[r * D_INNER + d];
    float xv  = xc[r * D_INNER + d];
    float Bv  = dbc[r * 64 + DT_RANK + n];
    float Cv  = dbc[r * 64 + DT_RANK + D_STATE + n];
    float dA  = __expf(dlt * a);
    h = dA * h + dlt * Bv * xv;
    float yv = h * Cv;
    yv += __shfl_xor(yv, 1, 16);
    yv += __shfl_xor(yv, 2, 16);
    yv += __shfl_xor(yv, 4, 16);
    yv += __shfl_xor(yv, 8, 16);
    if (n == 0) y[r * D_INNER + d] = yv + Dp * xv;
  }
}

// g = bf16(y * silu(z)),  z = xz[:, D_INNER:2*D_INNER]
__global__ __launch_bounds__(256) void gate_kernel(
    const float* __restrict__ y, const float* __restrict__ xz,
    unsigned short* __restrict__ g)
{
  int i = blockIdx.x * 256 + threadIdx.x;
  if (i >= NROWS * D_INNER) return;
  int row = i >> 10;
  int c   = i & (D_INNER - 1);
  float z = xz[(size_t)row * (2 * D_INNER) + D_INNER + c];
  float sig = 1.f / (1.f + __expf(-z));
  g[i] = f2bf(y[i] * z * sig);
}

__global__ __launch_bounds__(256) void f32_to_bf16_kernel(
    const float* __restrict__ in, unsigned short* __restrict__ out, int n)
{
  int i = blockIdx.x * 256 + threadIdx.x;
  if (i < n) out[i] = f2bf(in[i]);
}

// ======================================================================
extern "C" void kernel_launch(void* const* d_in, const int* in_sizes, int n_in,
                              void* d_out, int out_size, void* d_ws, size_t ws_size,
                              hipStream_t stream)
{
  (void)in_sizes; (void)n_in; (void)out_size; (void)ws_size;
  const float* x       = (const float*)d_in[0];
  const float* n1_w    = (const float*)d_in[1];
  const float* n1_b    = (const float*)d_in[2];
  const float* mn_w    = (const float*)d_in[3];
  const float* mn_b    = (const float*)d_in[4];
  const float* in_w    = (const float*)d_in[5];
  const float* in_b    = (const float*)d_in[6];
  const float* conv_w  = (const float*)d_in[7];
  const float* conv_b  = (const float*)d_in[8];
  const float* xp_w    = (const float*)d_in[9];
  const float* dt_w    = (const float*)d_in[10];
  const float* dt_b    = (const float*)d_in[11];
  const float* A_log   = (const float*)d_in[12];
  const float* D_param = (const float*)d_in[13];
  const float* out_w   = (const float*)d_in[14];
  const float* out_b   = (const float*)d_in[15];
  const float* n2_w    = (const float*)d_in[16];
  const float* n2_b    = (const float*)d_in[17];
  const float* kn_w    = (const float*)d_in[18];
  const float* kn_b    = (const float*)d_in[19];
  const float* grid_in = (const float*)d_in[20];
  const float* spl_w   = (const float*)d_in[21];

  // ---- workspace carving (aliasing is stream-order safe) ----
  char* ws = (char*)d_ws;
  size_t off = 0;
  auto carve = [&](size_t bytes) -> char* {
    char* p = ws + off;
    off += (bytes + 255) & ~(size_t)255;
    return p;
  };
  unsigned short* w_in_bf  = (unsigned short*)carve((size_t)2 * D_INNER * DIMX * 2);
  unsigned short* w_xp_bf  = (unsigned short*)carve((size_t)(DT_RANK + 2 * D_STATE) * D_INNER * 2);
  unsigned short* w_dt_bf  = (unsigned short*)carve((size_t)D_INNER * DT_RANK * 2);
  unsigned short* w_out_bf = (unsigned short*)carve((size_t)DIMX * D_INNER * 2);
  unsigned short* w_spl_bf = (unsigned short*)carve((size_t)DIMX * DIMX * NGRIDS * 2);
  unsigned short* u_bf     = (unsigned short*)carve((size_t)NROWS * DIMX * 2);
  float* xz                = (float*)carve((size_t)NROWS * 2 * D_INNER * 4);  // 32MB, later reused as basis
  float* xc                = (float*)carve((size_t)NROWS * D_INNER * 4);
  unsigned short* xc_bf    = (unsigned short*)carve((size_t)NROWS * D_INNER * 2);
  float* dbc               = (float*)carve((size_t)NROWS * 64 * 4);
  unsigned short* dbc_bf   = (unsigned short*)carve((size_t)NROWS * 64 * 2);
  float* delta             = (float*)carve((size_t)NROWS * D_INNER * 4);      // later reused for g_bf
  float* ybuf              = (float*)carve((size_t)NROWS * D_INNER * 4);      // later reused for x1
  unsigned short* basis = (unsigned short*)xz;     // NROWS*4096*2 == NROWS*2048*4
  unsigned short* g_bf  = (unsigned short*)delta;  // 8MB into a dead 16MB buffer
  float* x1 = ybuf;                                // y is dead after gate_kernel

  const dim3 blk256(256), blk128(128);

  // 1) weights -> bf16 (L2-resident thereafter)
  f32_to_bf16_kernel<<<(2 * D_INNER * DIMX + 255) / 256, blk256, 0, stream>>>(in_w,  w_in_bf,  2 * D_INNER * DIMX);
  f32_to_bf16_kernel<<<((DT_RANK + 2 * D_STATE) * D_INNER + 255) / 256, blk256, 0, stream>>>(xp_w, w_xp_bf, (DT_RANK + 2 * D_STATE) * D_INNER);
  f32_to_bf16_kernel<<<(D_INNER * DT_RANK + 255) / 256, blk256, 0, stream>>>(dt_w,  w_dt_bf,  D_INNER * DT_RANK);
  f32_to_bf16_kernel<<<(DIMX * D_INNER + 255) / 256, blk256, 0, stream>>>(out_w, w_out_bf, DIMX * D_INNER);
  f32_to_bf16_kernel<<<(DIMX * DIMX * NGRIDS + 255) / 256, blk256, 0, stream>>>(spl_w, w_spl_bf, DIMX * DIMX * NGRIDS);

  // 2) u = LN(LN(x)) -> bf16
  double_ln_bf16_kernel<<<NROWS, blk128, 0, stream>>>(x, n1_w, n1_b, mn_w, mn_b, u_bf);

  // 3) xz = u @ in_w.T + in_b           (M=4096, N=2048, K=512)
  gemm_bf16_wmma<0, true, false, false><<<dim3((2 * D_INNER) / 64, NROWS / 128), blk128, 0, stream>>>(
      u_bf, DIMX, w_in_bf, DIMX, in_b, nullptr, xz, nullptr, 2 * D_INNER, DIMX);

  // 4) xc = silu(depthwise_conv(xm) + conv_b)  (f32 + bf16)
  conv_silu_kernel<<<(NROWS * D_INNER + 255) / 256, blk256, 0, stream>>>(xz, conv_w, conv_b, xc, xc_bf);

  // 5) dbc = xc @ xp_w.T                (M=4096, N=64, K=1024), bf16 copy for dt GEMM
  gemm_bf16_wmma<0, false, false, true><<<dim3(1, NROWS / 128), blk128, 0, stream>>>(
      xc_bf, D_INNER, w_xp_bf, D_INNER, nullptr, nullptr, dbc, dbc_bf, 64, D_INNER);

  // 6) delta = softplus(dbc[:, :32] @ dt_w.T + dt_b)   (M=4096, N=1024, K=32)
  gemm_bf16_wmma<1, true, false, false><<<dim3(D_INNER / 64, NROWS / 128), blk128, 0, stream>>>(
      dbc_bf, 64, w_dt_bf, DT_RANK, dt_b, nullptr, delta, nullptr, D_INNER, DT_RANK);

  // 7) selective scan -> y (includes +D*xc); 32768 lanes, 16-lane reductions
  ssm_scan_kernel<<<(SEQ_B * D_INNER * D_STATE) / 256, blk256, 0, stream>>>(
      delta, dbc, xc, A_log, D_param, ybuf);

  // 8) g = bf16(y * silu(z))
  gate_kernel<<<(NROWS * D_INNER + 255) / 256, blk256, 0, stream>>>(ybuf, xz, g_bf);

  // 9) x1 = x + g @ out_w.T + out_b     (M=4096, N=512, K=1024)
  gemm_bf16_wmma<0, true, true, false><<<dim3(DIMX / 64, NROWS / 128), blk128, 0, stream>>>(
      g_bf, D_INNER, w_out_bf, D_INNER, out_b, x, x1, nullptr, DIMX, D_INNER);

  // 10) basis = 1 - tanh^2((LN(LN(x1)) - grid)/0.33) -> bf16 (4096 x 4096)
  kan_basis_kernel<<<NROWS, blk128, 0, stream>>>(x1, n2_w, n2_b, kn_w, kn_b, grid_in, basis);

  // 11) out = x1 + basis @ spl_w.T      (M=4096, N=512, K=4096)
  gemm_bf16_wmma<0, false, true, false><<<dim3(DIMX / 64, NROWS / 128), blk128, 0, stream>>>(
      basis, DIMX * NGRIDS, w_spl_bf, DIMX * NGRIDS, nullptr, x1, (float*)d_out, nullptr,
      DIMX, DIMX * NGRIDS);
}
